// Layer_Averaged_GwR_14164802142580
// MI455X (gfx1250) — matrix operand order repaired and checked by reading, not compiled, not run on hardware
//
#include <hip/hip_runtime.h>

// ---------------------------------------------------------------------------
// Problem constants (from reference): B=8, N=1024, GI=1024, GH=GO=128
// ---------------------------------------------------------------------------
#define BB   8
#define NN   1024
#define GII  1024
#define GHH  128

typedef __attribute__((ext_vector_type(16))) __bf16 v16bf_t;
typedef __attribute__((ext_vector_type(8)))  float  v8f_t;

union AFrag { v16bf_t v; unsigned int u[8]; };
union CFrag { v8f_t  v; float        f[8]; };

__device__ __forceinline__ unsigned short f2bf(float f) {
    unsigned int u = __float_as_uint(f);
    u += 0x7FFFu + ((u >> 16) & 1u);          // round-to-nearest-even
    return (unsigned short)(u >> 16);
}

// ---------------------------------------------------------------------------
// Kernel 1: deg^{-1/2} per row.  One wave (32 lanes) per adjacency row.
// deg = 1 (self loop) + #{j != i : rpa[b,i,j] != 0}
// ---------------------------------------------------------------------------
__global__ void gcn_deg_rsqrt(const int* __restrict__ rpa,
                              float* __restrict__ dinv) {
    const int wave = threadIdx.x >> 5;
    const int lane = threadIdx.x & 31;
    const long rowg = (long)blockIdx.x * 8 + wave;     // in [0, B*N)
    const int  i    = (int)(rowg % NN);
    const int* rp   = rpa + rowg * NN;
    int cnt = 0;
    for (int j = lane; j < NN; j += 32)
        cnt += (rp[j] != 0 && j != i) ? 1 : 0;
    #pragma unroll
    for (int off = 16; off; off >>= 1)
        cnt += __shfl_xor(cnt, off, 32);
    if (lane == 0)
        dinv[rowg] = rsqrtf((float)(cnt + 1));
}

// ---------------------------------------------------------------------------
// Kernel 2: Ahat[b,i,j] = dinv_i * dinv_j * A_ij  (diag forced to 1) -> bf16
// ---------------------------------------------------------------------------
__global__ void gcn_build_ahat(const int* __restrict__ rpa,
                               const float* __restrict__ dinv,
                               unsigned short* __restrict__ ahat) {
    const long idx = (long)blockIdx.x * blockDim.x + threadIdx.x; // over B*N*N
    const long row = idx / NN;          // b*N + i
    const int  j   = (int)(idx % NN);
    const int  i   = (int)(row % NN);
    const float di = dinv[row];
    const float dj = dinv[row - i + j];
    float v;
    if (i == j)               v = di * di;
    else if (rpa[idx] != 0)   v = di * dj;
    else                      v = 0.0f;
    ahat[idx] = f2bf(v);
}

// ---------------------------------------------------------------------------
// Kernel 3: generic f32 -> bf16 conversion
// ---------------------------------------------------------------------------
__global__ void gcn_cvt_bf16(const float* __restrict__ src,
                             unsigned short* __restrict__ dst, long n) {
    const long i = (long)blockIdx.x * blockDim.x + threadIdx.x;
    if (i < n) dst[i] = f2bf(src[i]);
}

// ---------------------------------------------------------------------------
// Kernel 4: bf16 WMMA GEMM  C[M,128] = A[M,K] @ Bm[K,128]  (+bias, relu)
//   NW = 128 is compile-time: all 8 B-staging loads fold to immediate offsets
//   block = 256 threads = 8 waves; block tile = 128(M) x 16(N)
//   K-step = 128: one LDS stage + 4x v_wmma_f32_16x16x32_bf16 per barrier pair
//   outputs: bf16 H (next layer operand) and/or f32 0.25*H into `fout`
//   fmode: 0 = none, 1 = fout = 0.25*h, 2 = fout += 0.25*h
// ---------------------------------------------------------------------------
__global__ void gcn_gemm_bf16_wmma(const unsigned short* __restrict__ A, long sA,
                                   const unsigned short* __restrict__ Bm, long sB,
                                   const float* __restrict__ bias,
                                   unsigned short* __restrict__ Hout, long sH,
                                   float* __restrict__ Fout, long sF,
                                   int K, int do_relu, int fmode) {
    constexpr int NW = GHH;                    // output width, always 128
    // 16 cols x 128 K of bf16, padded to 65 dwords per col (bank-conflict-free)
    __shared__ unsigned int ldsB32[16 * 65];

    const int tid  = threadIdx.x;
    const int wave = tid >> 5;
    const int lane = tid & 31;
    const int hf   = lane >> 4;                // K-half selector (lanes 16-31)
    const int col  = lane & 15;                // output column within tile
    const int b    = blockIdx.z;
    const int tn   = blockIdx.x * 16;          // tile col base
    const int tm   = blockIdx.y * 128 + wave * 16;  // tile row base (this wave)

    // --- A fragment base: lanes 0-15 / 16-31 both map to rows 0..15 ---------
    const unsigned short* Aptr = A + (long)b * sA + (long)(tm + col) * K;
    const unsigned int*   A32  = (const unsigned int*)Aptr;   // advances 64/iter

    // --- B staging base: thread t covers (k = k0+16s, c) for s = 0..7 -------
    const int k0 = tid >> 4;                   // 0..15
    const int c0 = tid & 15;
    const unsigned short* Bst =
        Bm + (long)b * sB + tn + c0 + k0 * NW;                // advances 128*NW/iter
    unsigned short* lds16 = (unsigned short*)ldsB32;
    unsigned short* const ldsW = lds16 + c0 * 130 + k0;       // 130 us = 65 dw stride

    CFrag acc;
    #pragma unroll
    for (int r = 0; r < 8; ++r) acc.f[r] = 0.0f;

    const int nkt = K >> 7;                    // K / 128 iterations
    for (int it = 0; it < nkt; ++it) {
        __syncthreads();
        // stage B panel [128 x 16] -> LDS (col-major, pair-contiguous K);
        // s * 16 * NW is a compile-time byte offset -> immediate VMEM offsets
        #pragma unroll
        for (int s = 0; s < 8; ++s)
            ldsW[16 * s] = Bst[s * (16 * NW)];
        Bst += 128 * NW;
        __syncthreads();

        if (it + 1 < nkt)
            __builtin_prefetch((const unsigned short*)A32 + 128, 0, 3); // global_prefetch_b8

        #pragma unroll
        for (int t = 0; t < 4; ++t) {          // four K=32 WMMA steps
            AFrag a, bfr;
            #pragma unroll
            for (int g = 0; g < 8; ++g)        // 16-bit A 16x32 VGPR layout
                a.u[g] = A32[t * 16 + hf * 8 + g];
            #pragma unroll
            for (int g = 0; g < 8; ++g)        // 16-bit B 32x16 VGPR layout
                bfr.u[g] = ldsB32[col * 65 + t * 16 + hf * 8 + g];
            acc.v = __builtin_amdgcn_wmma_f32_16x16x32_bf16(
                false, a.v, false, bfr.v, (short)0, acc.v, false, false);
        }
        A32 += 64;                              // 128 bf16 = 64 dwords
    }

    // Epilogue: bias + relu, bf16 requantize for next layer, 0.25*H into fout
    const float bcol = bias ? bias[tn + col] : 0.0f;
    unsigned short* Hp = Hout ? Hout + (long)b * sH : nullptr;
    float*          Fp = (fmode && Fout) ? Fout + (long)b * sF : nullptr;
    #pragma unroll
    for (int r = 0; r < 8; ++r) {
        const int gr = tm + hf * 8 + r;        // C/D layout: VGPR r -> M = r + 8*hf
        float h = acc.f[r] + bcol;
        if (do_relu) h = fmaxf(h, 0.0f);
        const long o = (long)gr * NW + tn + col;
        if (Hp) Hp[o] = f2bf(h);
        if (fmode == 1)      Fp[o]  = 0.25f * h;
        else if (fmode == 2) Fp[o] += 0.25f * h;
    }
}

// ---------------------------------------------------------------------------
// Host-side orchestration
// ---------------------------------------------------------------------------
extern "C" void kernel_launch(void* const* d_in, const int* in_sizes, int n_in,
                              void* d_out, int out_size, void* d_ws, size_t ws_size,
                              hipStream_t stream) {
    (void)in_sizes; (void)n_in; (void)out_size; (void)ws_size;

    const float* aa    = (const float*)d_in[0];   // [B,N,GI]
    const int*   rpa   = (const int*)  d_in[1];   // [B,N,N]
    const float* W_in  = (const float*)d_in[2];   // [GI,GH]
    const float* b_in  = (const float*)d_in[3];
    const float* W_h0  = (const float*)d_in[4];   // [GH,GH]
    const float* b_h0  = (const float*)d_in[5];
    const float* W_h1  = (const float*)d_in[6];
    const float* b_h1  = (const float*)d_in[7];
    const float* W_out = (const float*)d_in[8];
    const float* b_out = (const float*)d_in[9];
    float* out = (float*)d_out;                   // [B,N,GH]

    // workspace carve-up (256B aligned)
    char* ws = (char*)d_ws;
    size_t off = 0;
    auto carve = [&](size_t bytes) { void* p = ws + off; off = (off + bytes + 255) & ~(size_t)255; return p; };
    float*          dinv  = (float*)         carve((size_t)BB * NN * 4);
    unsigned short* ahat  = (unsigned short*)carve((size_t)BB * NN * NN * 2);
    unsigned short* xbf   = (unsigned short*)carve((size_t)BB * NN * GII * 2);
    unsigned short* winb  = (unsigned short*)carve((size_t)GII * GHH * 2);
    unsigned short* w0b   = (unsigned short*)carve((size_t)GHH * GHH * 2);
    unsigned short* w1b   = (unsigned short*)carve((size_t)GHH * GHH * 2);
    unsigned short* wob   = (unsigned short*)carve((size_t)GHH * GHH * 2);
    unsigned short* Tb    = (unsigned short*)carve((size_t)BB * NN * GHH * 2);
    unsigned short* Ha    = (unsigned short*)carve((size_t)BB * NN * GHH * 2);
    unsigned short* Hb    = (unsigned short*)carve((size_t)BB * NN * GHH * 2);

    const long sAhat = (long)NN * NN;
    const long sX    = (long)NN * GII;
    const long sH    = (long)NN * GHH;

    // 1) degree -> deg^{-1/2}
    gcn_deg_rsqrt<<<(BB * NN) / 8, 256, 0, stream>>>(rpa, dinv);
    // 2) normalized adjacency in bf16
    gcn_build_ahat<<<(BB * NN * NN) / 256, 256, 0, stream>>>(rpa, dinv, ahat);
    // 3) bf16 conversions
    gcn_cvt_bf16<<<(BB * NN * GII) / 256, 256, 0, stream>>>(aa, xbf, (long)BB * NN * GII);
    gcn_cvt_bf16<<<(GII * GHH) / 256, 256, 0, stream>>>(W_in, winb, (long)GII * GHH);
    gcn_cvt_bf16<<<(GHH * GHH) / 256, 256, 0, stream>>>(W_h0, w0b, (long)GHH * GHH);
    gcn_cvt_bf16<<<(GHH * GHH) / 256, 256, 0, stream>>>(W_h1, w1b, (long)GHH * GHH);
    gcn_cvt_bf16<<<(GHH * GHH) / 256, 256, 0, stream>>>(W_out, wob, (long)GHH * GHH);

    const dim3 gg(GHH / 16, NN / 128, BB);  // (8, 8, 8)
    const dim3 bl(256, 1, 1);

    // ----- layer_in:  T = X @ W_in ; H1 = relu(Ahat@T + b_in); out = 0.25*H1
    gcn_gemm_bf16_wmma<<<gg, bl, 0, stream>>>(xbf, sX, winb, 0, nullptr,
                                              Tb, sH, nullptr, 0,
                                              GII, 0, 0);
    gcn_gemm_bf16_wmma<<<gg, bl, 0, stream>>>(ahat, sAhat, Tb, sH, b_in,
                                              Ha, sH, out, sH,
                                              NN, 1, 1);
    // ----- hidden 0:  T = H1 @ W_h0 ; H2 = relu(Ahat@T + b_h0); out += 0.25*H2
    gcn_gemm_bf16_wmma<<<gg, bl, 0, stream>>>(Ha, sH, w0b, 0, nullptr,
                                              Tb, sH, nullptr, 0,
                                              GHH, 0, 0);
    gcn_gemm_bf16_wmma<<<gg, bl, 0, stream>>>(ahat, sAhat, Tb, sH, b_h0,
                                              Hb, sH, out, sH,
                                              NN, 1, 2);
    // ----- hidden 1:  T = H2 @ W_h1 ; H3 = relu(Ahat@T + b_h1); out += 0.25*H3
    gcn_gemm_bf16_wmma<<<gg, bl, 0, stream>>>(Hb, sH, w1b, 0, nullptr,
                                              Tb, sH, nullptr, 0,
                                              GHH, 0, 0);
    gcn_gemm_bf16_wmma<<<gg, bl, 0, stream>>>(ahat, sAhat, Tb, sH, b_h1,
                                              Ha, sH, out, sH,
                                              NN, 1, 2);
    // ----- layer_out: T = H3 @ W_out ; H4 = relu(Ahat@T + b_out); out += 0.25*H4
    gcn_gemm_bf16_wmma<<<gg, bl, 0, stream>>>(Ha, sH, wob, 0, nullptr,
                                              Tb, sH, nullptr, 0,
                                              GHH, 0, 0);
    gcn_gemm_bf16_wmma<<<gg, bl, 0, stream>>>(ahat, sAhat, Tb, sH, b_out,
                                              nullptr, 0, out, sH,
                                              NN, 1, 2);
}
